// STTG_49185965474232
// MI455X (gfx1250) — matrix operand order repaired and checked by reading, not compiled, not run on hardware
//
#include <hip/hip_runtime.h>
#include <hip/hip_bf16.h>
#include <math.h>

// ---------------------------------------------------------------------------
// Problem constants (from reference setup_inputs)
// ---------------------------------------------------------------------------
constexpr int NB  = 2;        // batch
constexpr int C3  = 256;      // level-3 channels
constexpr int Hc  = 48;
constexpr int Wc  = 48;
constexpr int HWc = Hc * Wc;        // 2304
constexpr int K9c = C3 * 9;         // 2304 (unfold 3x3 of 256ch)
constexpr int MAXK = 5;

typedef __bf16 bf16_t;
typedef __attribute__((ext_vector_type(16))) __bf16 v16bf;
typedef __attribute__((ext_vector_type(8)))  __bf16 v8bf;
typedef __attribute__((ext_vector_type(8)))  float  v8f;

__device__ __forceinline__ unsigned short f2bf(float f) {
    union { float f; unsigned u; } v; v.f = f;
    unsigned r = v.u + 0x7FFFu + ((v.u >> 16) & 1u);   // RNE
    return (unsigned short)(r >> 16);
}

// ---------------------------------------------------------------------------
// Kernel 1: positional encoding  out = x + conv3x3(posgrid, w) + b
// posgrid is bilinear-resized meshgrid of linspace(-1,1,ms): separable.
// ---------------------------------------------------------------------------
__global__ __launch_bounds__(256)
void posenc_kernel(const float* __restrict__ x, const float* __restrict__ w,
                   const float* __restrict__ b, float* __restrict__ out,
                   int C, int Hh, int Ww, int ms) {
    size_t total = (size_t)NB * C * Hh * Ww;
    size_t idx = (size_t)blockIdx.x * 256 + threadIdx.x;
    if (idx >= total) return;
    int xx = (int)(idx % Ww); size_t t1 = idx / Ww;
    int yy = (int)(t1 % Hh); size_t t2 = t1 / Hh;
    int c  = (int)(t2 % C);

    auto pg = [&](int coord, int extent) -> float {
        float src = ((float)coord + 0.5f) * ((float)ms / (float)extent) - 0.5f;
        src = fminf(fmaxf(src, 0.f), (float)ms - 1.f);
        int i0 = (int)floorf(src);
        int i1 = min(i0 + 1, ms - 1);
        float f = src - (float)i0;
        float step = 2.0f / (float)(ms - 1);
        float l0 = -1.0f + step * (float)i0;
        float l1 = -1.0f + step * (float)i1;
        return l0 * (1.f - f) + l1 * f;
    };

    float acc = b[c];
    for (int dy = 0; dy < 3; ++dy) {
        int sy = yy + dy - 1;
        if (sy < 0 || sy >= Hh) continue;
        float p0 = pg(sy, Hh);
        for (int dx = 0; dx < 3; ++dx) {
            int sx = xx + dx - 1;
            if (sx < 0 || sx >= Ww) continue;
            float p1 = pg(sx, Ww);
            acc += w[((c * 2 + 0) * 3 + dy) * 3 + dx] * p0
                 + w[((c * 2 + 1) * 3 + dy) * 3 + dx] * p1;
        }
    }
    out[idx] = x[idx] + acc;
}

// ---------------------------------------------------------------------------
// Kernel 2: unfold(3x3,pad1) + L2-normalize per spatial position -> bf16 row
// dst[n][pos][k], k = c*9 + i*3 + j   (same ordering both sides => dot OK)
// ---------------------------------------------------------------------------
__global__ __launch_bounds__(128)
void unfnorm_kernel(const float* __restrict__ src, unsigned short* __restrict__ dst) {
    int bp  = blockIdx.x;
    int n   = bp / HWc, pos = bp % HWc;
    int py  = pos / Wc, px = pos % Wc;
    const float* s = src + (size_t)n * C3 * HWc;

    float ss = 0.f;
    for (int k = threadIdx.x; k < K9c; k += 128) {
        int c = k / 9, r = k % 9, di = r / 3, dj = r % 3;
        int y = py + di - 1, x = px + dj - 1;
        float v = (y >= 0 && y < Hc && x >= 0 && x < Wc)
                    ? s[(size_t)c * HWc + y * Wc + x] : 0.f;
        ss += v * v;
    }
    __shared__ float red[128];
    red[threadIdx.x] = ss; __syncthreads();
    for (int st = 64; st > 0; st >>= 1) {
        if ((int)threadIdx.x < st) red[threadIdx.x] += red[threadIdx.x + st];
        __syncthreads();
    }
    float inv = 1.0f / fmaxf(sqrtf(red[0]), 1e-12f);

    unsigned short* d = dst + ((size_t)n * HWc + pos) * K9c;
    for (int k = threadIdx.x; k < K9c; k += 128) {
        int c = k / 9, r = k % 9, di = r / 3, dj = r % 3;
        int y = py + di - 1, x = px + dj - 1;
        float v = (y >= 0 && y < Hc && x >= 0 && x < Wc)
                    ? s[(size_t)c * HWc + y * Wc + x] : 0.f;
        d[k] = f2bf(v * inv);
    }
}

// ---------------------------------------------------------------------------
// Kernel 3: global cosine similarity scalar per batch
// ---------------------------------------------------------------------------
__global__ __launch_bounds__(256)
void gsim_kernel(const float* __restrict__ encref, const float* __restrict__ lrsr,
                 float* __restrict__ Rg) {
    int n = blockIdx.x;
    int c = threadIdx.x;     // 256 channels
    const float* a = encref + ((size_t)n * C3 + c) * HWc;
    const float* b = lrsr   + ((size_t)n * C3 + c) * HWc;
    float ma = 0.f, mb = 0.f;
    for (int i = 0; i < HWc; ++i) { ma += a[i]; mb += b[i]; }
    ma /= (float)HWc; mb /= (float)HWc;
    __shared__ float sa[256], sb[256], sd[256];
    sa[c] = ma * ma; sb[c] = mb * mb; sd[c] = ma * mb;
    __syncthreads();
    for (int st = 128; st > 0; st >>= 1) {
        if (c < st) { sa[c] += sa[c + st]; sb[c] += sb[c + st]; sd[c] += sd[c + st]; }
        __syncthreads();
    }
    if (c == 0)
        Rg[n] = sd[0] / (fmaxf(sqrtf(sa[0]), 1e-12f) * fmaxf(sqrtf(sb[0]), 1e-12f));
}

// ---------------------------------------------------------------------------
// Kernel 4: WMMA bf16 GEMM with fused global-similarity blend.
// A: [n][q][k] bf16 (refsr rows), B: [n][p][k] bf16 (lrsr cols, stored as rows).
// RT[n][p][q] = (1-gw) * dot(A[q],B[p]) + gw * Rglob[n]   (transposed store!)
// Block: 128x128 tile, 256 thr = 8 waves, wave tile 64x32 (4x2 WMMA accs).
// Global->LDS via gfx1250 GLOBAL_LOAD_ASYNC_TO_LDS_B128 (ASYNCcnt-tracked,
// no VGPR staging), double-buffered so the prefetch of tile k+1 overlaps
// the 8 WMMAs consuming tile k.
// ---------------------------------------------------------------------------
#define LDSW 40   // bf16 elems per LDS row (32 + 8 pad)

__global__ __launch_bounds__(256)
void gemm_wmma_kernel(const unsigned short* __restrict__ A,
                      const unsigned short* __restrict__ B,
                      float* __restrict__ RT,
                      const float* __restrict__ Rglob,
                      const float* __restrict__ gw_ptr) {
    __shared__ __align__(16) unsigned short sA[2][128 * LDSW];
    __shared__ __align__(16) unsigned short sB[2][128 * LDSW];

    const int n       = blockIdx.z;
    const int rowBase = blockIdx.y * 128;   // q (ref) tile
    const int colBase = blockIdx.x * 128;   // p (lr)  tile
    const int tid  = threadIdx.x;
    const int lane = tid & 31;
    const int wave = tid >> 5;
    const int wm   = wave & 1;    // M: 2 waves * 64
    const int wn   = wave >> 1;   // N: 4 waves * 32
    const int half = lane >> 4;
    const int l16  = lane & 15;

    const unsigned short* Abase = A + ((size_t)n * HWc + rowBase) * K9c;
    const unsigned short* Bbase = B + ((size_t)n * HWc + colBase) * K9c;

    v8f acc[4][2];
    for (int i = 0; i < 4; ++i)
        for (int j = 0; j < 2; ++j)
            for (int e = 0; e < 8; ++e) acc[i][j][e] = 0.f;

    const int ldr = tid >> 1;            // row 0..127 this thread copies
    const int ldc = (tid & 1) * 16;      // 16-elem half of the 32-elem k slab
    const unsigned gofs0 = (unsigned)((ldr * K9c + ldc) * 2);   // bytes

    // Issue 4 async b128 copies: two 16B chunks of A and of B for this thread.
    // instruction offset applies to BOTH the LDS and global address (ISA 10.x),
    // so one address VGPR per matrix covers both chunks.
    auto issue = [&](int buf, int k0) {
        unsigned la = (unsigned)(size_t)&sA[buf][ldr * LDSW + ldc];
        unsigned lb = (unsigned)(size_t)&sB[buf][ldr * LDSW + ldc];
        unsigned go = gofs0 + (unsigned)(k0 * 2);
        asm volatile(
            "global_load_async_to_lds_b128 %0, %2, %3\n\t"
            "global_load_async_to_lds_b128 %0, %2, %3 offset:16\n\t"
            "global_load_async_to_lds_b128 %1, %2, %4\n\t"
            "global_load_async_to_lds_b128 %1, %2, %4 offset:16"
            :: "v"(la), "v"(lb), "v"(go), "s"(Abase), "s"(Bbase)
            : "memory");
    };

    issue(0, 0);                 // prologue prefetch
    int cur = 0;
    for (int k0 = 0; k0 < K9c; k0 += 32) {
        asm volatile("s_wait_asynccnt 0x0" ::: "memory");
        __syncthreads();                              // tile `cur` resident
        if (k0 + 32 < K9c) issue(cur ^ 1, k0 + 32);   // overlap next prefetch

        const unsigned short* aP = sA[cur];
        const unsigned short* bP = sB[cur];

        // A fragments: lane half h -> K {h*8..h*8+7} and {16+h*8..16+h*8+7}
        v16bf afr[4];
        for (int sm = 0; sm < 4; ++sm) {
            int row = wm * 64 + sm * 16 + l16;
            v8bf lo = *(const v8bf*)(aP + row * LDSW + half * 8);
            v8bf hi = *(const v8bf*)(aP + row * LDSW + 16 + half * 8);
            for (int e = 0; e < 8; ++e) { afr[sm][e] = lo[e]; afr[sm][8 + e] = hi[e]; }
        }
        // B fragments: lane half h -> K {h*8..h*8+7} and {16+h*8..16+h*8+7}
        v16bf bfr[2];
        for (int sn = 0; sn < 2; ++sn) {
            int row = wn * 32 + sn * 16 + l16;
            v8bf lo = *(const v8bf*)(bP + row * LDSW + half * 8);
            v8bf hi = *(const v8bf*)(bP + row * LDSW + 16 + half * 8);
            for (int e = 0; e < 8; ++e) { bfr[sn][e] = lo[e]; bfr[sn][8 + e] = hi[e]; }
        }
        for (int sm = 0; sm < 4; ++sm)
            for (int sn = 0; sn < 2; ++sn)
                acc[sm][sn] = __builtin_amdgcn_wmma_f32_16x16x32_bf16(
                    false, afr[sm], false, bfr[sn], (short)0, acc[sm][sn], false, false);

        __syncthreads();                              // done reading `cur`
        cur ^= 1;
    }

    const float gw = *gw_ptr;
    const float addg = gw * Rglob[n];
    const float scl = 1.0f - gw;

    for (int sm = 0; sm < 4; ++sm) {
        int qb = rowBase + wm * 64 + sm * 16 + half * 8;   // base q for this lane
        for (int sn = 0; sn < 2; ++sn) {
            int pcol = colBase + wn * 32 + sn * 16 + l16;
            float* out = RT + ((size_t)n * HWc + pcol) * HWc + qb;
            for (int v = 0; v < 8; ++v)                    // m = v + half*8
                out[v] = scl * acc[sm][sn][v] + addg;
        }
    }
}

// ---------------------------------------------------------------------------
// Kernel 5: top-5 per RT row (= per lr position, over ref positions).
// One wave per row; lane-local insertion sort, lane-0 merge through LDS.
// ---------------------------------------------------------------------------
__device__ __forceinline__ void top5_ins(float* v, int* ix, float x, int q) {
    if (x > v[4] || (x == v[4] && q < ix[4])) {
        int t = 4;
        while (t > 0 && (x > v[t - 1] || (x == v[t - 1] && q < ix[t - 1]))) {
            v[t] = v[t - 1]; ix[t] = ix[t - 1]; --t;
        }
        v[t] = x; ix[t] = q;
    }
}

__global__ __launch_bounds__(256)
void topk_kernel(const float* __restrict__ RT, float* __restrict__ topv,
                 int* __restrict__ topi) {
    int w    = threadIdx.x >> 5;
    int lane = threadIdx.x & 31;
    int row  = blockIdx.x * 8 + w;          // n*HW + p  (grid sized exactly)
    const float* r = RT + (size_t)row * HWc;

    float v[5]; int ix[5];
    for (int t = 0; t < 5; ++t) { v[t] = -INFINITY; ix[t] = 0x7fffffff; }
    for (int q = lane; q < HWc; q += 32) top5_ins(v, ix, r[q], q);

    __shared__ float sv[8][32][5];
    __shared__ int   si[8][32][5];
    for (int t = 0; t < 5; ++t) { sv[w][lane][t] = v[t]; si[w][lane][t] = ix[t]; }
    __syncthreads();

    if (lane == 0) {
        float bv[5]; int bi[5];
        for (int t = 0; t < 5; ++t) { bv[t] = -INFINITY; bi[t] = 0x7fffffff; }
        for (int l = 0; l < 32; ++l)
            for (int t = 0; t < 5; ++t) top5_ins(bv, bi, sv[w][l][t], si[w][l][t]);
        for (int t = 0; t < 5; ++t) {
            topv[(size_t)row * 5 + t] = bv[t];
            topi[(size_t)row * 5 + t] = bi[t];
        }
    }
}

// ---------------------------------------------------------------------------
// Kernel 6: Gumbel-softmax hard k selection (threefry2x32, JAX key(42)).
// ---------------------------------------------------------------------------
__device__ void threefry2x32(unsigned k0, unsigned k1, unsigned x0, unsigned x1,
                             unsigned* o0, unsigned* o1) {
    const unsigned rot[8] = {13, 15, 26, 6, 17, 29, 16, 24};
    unsigned ks[3] = {k0, k1, k0 ^ k1 ^ 0x1BD11BDAu};
    unsigned a = x0 + ks[0], b = x1 + ks[1];
    for (int d = 0; d < 5; ++d) {
        for (int r = 0; r < 4; ++r) {
            unsigned R = rot[(d & 1) * 4 + r];
            a += b; b = (b << R) | (b >> (32 - R)); b ^= a;
        }
        a += ks[(d + 1) % 3];
        b += ks[(d + 2) % 3] + (unsigned)(d + 1);
    }
    *o0 = a; *o1 = b;
}

__global__ void kselect_kernel(const float* __restrict__ k_logits,
                               const float* __restrict__ temp,
                               int* __restrict__ khard) {
    if (threadIdx.x != 0 || blockIdx.x != 0) return;
    unsigned o0[8], o1[8];
    for (int i = 0; i < 8; ++i) {
        unsigned x1 = (i < 7) ? (unsigned)(8 + i) : 0u;   // counts padded with 0
        threefry2x32(0u, 42u, (unsigned)i, x1, &o0[i], &o1[i]);
    }
    float g[15];
    for (int j = 0; j < 15; ++j) {
        unsigned bits = (j < 8) ? o0[j] : o1[j - 8];
        union { unsigned u; float f; } cv; cv.u = (bits >> 9) | 0x3f800000u;
        float u01 = cv.f - 1.0f;                                 // [0,1)
        float u = fmaxf(u01 * (1.0f - 1e-6f) + 1e-6f, 1e-6f);    // [1e-6,1)
        g[j] = -logf(-logf(u));
    }
    float T = *temp;
    for (int lvl = 0; lvl < 3; ++lvl) {
        float best = -INFINITY; int bi = 0;
        for (int t = 0; t < 5; ++t) {
            float z = (k_logits[lvl * 5 + t] + g[lvl * 5 + t]) / T;   // argmax==softmax argmax
            if (z > best) { best = z; bi = t; }
        }
        khard[lvl] = bi + 1;
    }
}

// ---------------------------------------------------------------------------
// Kernel 7: per-level masked sigmoid weights: wsel[lvl][n][t][p]
// ---------------------------------------------------------------------------
__global__ void wsel_kernel(const float* __restrict__ topv,
                            const int* __restrict__ khard,
                            float* __restrict__ wsel) {
    int idx = blockIdx.x * 256 + threadIdx.x;
    if (idx >= 3 * NB * HWc) return;
    int lvl = idx / (NB * HWc);
    int rem = idx % (NB * HWc);
    int n = rem / HWc, p = rem % HWc;
    int kh = khard[lvl];
    for (int t = 0; t < 5; ++t) {
        float v = topv[((size_t)n * HWc + p) * 5 + t];
        float w = (t < kh) ? 1.0f / (1.0f + expf(-v)) : 0.0f;
        wsel[(((size_t)lvl * NB + n) * 5 + t) * HWc + p] = w;
    }
}

// ---------------------------------------------------------------------------
// Kernel 8: transfer = fold(sum_t sigmoid(R)*gathered unfold patches)/k^2.
// Output-centric: each output pixel sums its <= (k/s)^2=9 overlapping patches.
// ---------------------------------------------------------------------------
__global__ __launch_bounds__(256)
void transfer_kernel(const float* __restrict__ enc, const float* __restrict__ wsel,
                     const int* __restrict__ topi, float* __restrict__ out,
                     int C, int s, int kern, int pad) {
    const int OH = Hc * s, OW = Wc * s;
    size_t total = (size_t)NB * C * OH * OW;
    size_t idx = (size_t)blockIdx.x * 256 + threadIdx.x;
    if (idx >= total) return;
    int ox = (int)(idx % OW); size_t t1 = idx / OW;
    int oy = (int)(t1 % OH); size_t t2 = t1 / OH;
    int c  = (int)(t2 % C);  int n = (int)(t2 / C);
    const float* e = enc + ((size_t)n * C + c) * (size_t)OH * OW;

    float acc = 0.f;
    int pyc = (oy + pad) / s, pxc = (ox + pad) / s;
    int span = kern / s;     // 3
    for (int dy = 0; dy <= span; ++dy) {
        int py = pyc - dy; if (py < 0 || py >= Hc) continue;
        int i = oy + pad - py * s; if (i < 0 || i >= kern) continue;
        for (int dx = 0; dx <= span; ++dx) {
            int px = pxc - dx; if (px < 0 || px >= Wc) continue;
            int j = ox + pad - px * s; if (j < 0 || j >= kern) continue;
            int p = py * Wc + px;
            for (int t = 0; t < 5; ++t) {
                float w = wsel[((size_t)n * 5 + t) * HWc + p];
                if (w == 0.f) continue;
                int r  = topi[((size_t)n * HWc + p) * 5 + t];
                int ry = r / Wc, rx = r % Wc;
                int sy = ry * s + i - pad, sx = rx * s + j - pad;
                float v = (sy >= 0 && sy < OH && sx >= 0 && sx < OW)
                            ? e[(size_t)sy * OW + sx] : 0.f;
                acc += w * v;
            }
        }
    }
    out[idx] = acc / (float)(kern * kern);
}

// ---------------------------------------------------------------------------
// Kernel 9/10: Squeeze-Excite (fc chain then in-place scale of d_out region)
// ---------------------------------------------------------------------------
__global__ __launch_bounds__(256)
void se_fc_kernel(const float* __restrict__ T, const float* __restrict__ w1,
                  const float* __restrict__ b1, const float* __restrict__ w2,
                  const float* __restrict__ b2, float* __restrict__ scale,
                  int C, int spatial) {
    int n = blockIdx.x;
    int R = C / 16;
    __shared__ float mean_s[256];
    __shared__ float h_s[16];
    if ((int)threadIdx.x < C) {
        const float* p = T + ((size_t)n * C + threadIdx.x) * spatial;
        float m = 0.f;
        for (int i = 0; i < spatial; ++i) m += p[i];
        mean_s[threadIdx.x] = m / (float)spatial;
    }
    __syncthreads();
    if ((int)threadIdx.x < R) {
        float a = b1[threadIdx.x];
        for (int c = 0; c < C; ++c) a += w1[threadIdx.x * C + c] * mean_s[c];
        h_s[threadIdx.x] = fmaxf(a, 0.f);
    }
    __syncthreads();
    if ((int)threadIdx.x < C) {
        float a = b2[threadIdx.x];
        for (int r = 0; r < R; ++r) a += w2[threadIdx.x * R + r] * h_s[r];
        scale[(size_t)n * C + threadIdx.x] = 1.0f / (1.0f + expf(-a));
    }
}

__global__ void se_scale_kernel(float* __restrict__ T, const float* __restrict__ scale,
                                int C, int spatial) {
    size_t total = (size_t)NB * C * spatial;
    size_t idx = (size_t)blockIdx.x * 256 + threadIdx.x;
    if (idx >= total) return;
    int c = (int)((idx / spatial) % C);
    int n = (int)(idx / ((size_t)spatial * C));
    T[idx] *= scale[(size_t)n * C + c];
}

// ---------------------------------------------------------------------------
// Host orchestration
// ---------------------------------------------------------------------------
extern "C" void kernel_launch(void* const* d_in, const int* in_sizes, int n_in,
                              void* d_out, int out_size, void* d_ws, size_t ws_size,
                              hipStream_t stream) {
    (void)in_sizes; (void)n_in; (void)out_size; (void)ws_size;
    const float* lrsr  = (const float*)d_in[0];
    const float* refsr = (const float*)d_in[1];
    const float* ref1  = (const float*)d_in[2];
    const float* ref2  = (const float*)d_in[3];
    const float* ref3  = (const float*)d_in[4];
    const float* klog  = (const float*)d_in[5];
    const float* temp  = (const float*)d_in[6];
    const float* gw    = (const float*)d_in[7];
    const float* pw1 = (const float*)d_in[8],  *pb1 = (const float*)d_in[9];
    const float* pw2 = (const float*)d_in[10], *pb2 = (const float*)d_in[11];
    const float* pw3 = (const float*)d_in[12], *pb3 = (const float*)d_in[13];
    const float* se1w1 = (const float*)d_in[14], *se1b1 = (const float*)d_in[15];
    const float* se1w2 = (const float*)d_in[16], *se1b2 = (const float*)d_in[17];
    const float* se2w1 = (const float*)d_in[18], *se2b1 = (const float*)d_in[19];
    const float* se2w2 = (const float*)d_in[20], *se2b2 = (const float*)d_in[21];
    const float* se3w1 = (const float*)d_in[22], *se3b1 = (const float*)d_in[23];
    const float* se3w2 = (const float*)d_in[24], *se3b2 = (const float*)d_in[25];

    // workspace carve-out (256B aligned slices)
    size_t off = 0;
    auto carve = [&](size_t bytes) -> void* {
        void* p = (char*)d_ws + off;
        off += (bytes + 255) & ~(size_t)255;
        return p;
    };
    float* enc_refsr = (float*)carve((size_t)NB * C3 * HWc * 4);
    float* enc_ref3  = (float*)carve((size_t)NB * C3 * HWc * 4);
    float* enc_ref2  = (float*)carve((size_t)NB * 128 * 96 * 96 * 4);
    float* enc_ref1  = (float*)carve((size_t)NB * 64 * 192 * 192 * 4);
    unsigned short* Abf = (unsigned short*)carve((size_t)NB * HWc * K9c * 2);
    unsigned short* Bbf = (unsigned short*)carve((size_t)NB * HWc * K9c * 2);
    float* RT    = (float*)carve((size_t)NB * HWc * HWc * 4);
    float* Rglob = (float*)carve(NB * 4);
    float* topv  = (float*)carve((size_t)NB * HWc * 5 * 4);
    int*   topi  = (int*)  carve((size_t)NB * HWc * 5 * 4);
    int*   khard = (int*)  carve(3 * 4);
    float* wsel  = (float*)carve((size_t)3 * NB * 5 * HWc * 4);
    float* sesc3 = (float*)carve((size_t)NB * 256 * 4);
    float* sesc2 = (float*)carve((size_t)NB * 128 * 4);
    float* sesc1 = (float*)carve((size_t)NB * 64 * 4);

    float* T3 = (float*)d_out;                                    // 2*256*48*48
    float* T2 = T3 + (size_t)NB * 256 * HWc;                      // 2*128*96*96
    float* T1 = T2 + (size_t)NB * 128 * 96 * 96;                  // 2*64*192*192

    auto blk = [](size_t total) { return (unsigned)((total + 255) / 256); };

    // 1) positional encodings
    posenc_kernel<<<blk((size_t)NB*C3*HWc), 256, 0, stream>>>(refsr, pw3, pb3, enc_refsr, 256, 48, 48, 64);
    posenc_kernel<<<blk((size_t)NB*C3*HWc), 256, 0, stream>>>(ref3,  pw3, pb3, enc_ref3,  256, 48, 48, 64);
    posenc_kernel<<<blk((size_t)NB*128*96*96), 256, 0, stream>>>(ref2, pw2, pb2, enc_ref2, 128, 96, 96, 128);
    posenc_kernel<<<blk((size_t)NB*64*192*192), 256, 0, stream>>>(ref1, pw1, pb1, enc_ref1, 64, 192, 192, 256);

    // 2) unfold + normalize -> bf16 operands
    unfnorm_kernel<<<NB * HWc, 128, 0, stream>>>(enc_refsr, Abf);
    unfnorm_kernel<<<NB * HWc, 128, 0, stream>>>(lrsr,      Bbf);

    // 3) global cosine similarity
    gsim_kernel<<<NB, 256, 0, stream>>>(enc_refsr, lrsr, Rglob);

    // 4) WMMA GEMM with fused blend, transposed store (async double-buffered)
    gemm_wmma_kernel<<<dim3(HWc/128, HWc/128, NB), 256, 0, stream>>>(Abf, Bbf, RT, Rglob, gw);

    // 5) top-5 per lr position
    topk_kernel<<<(NB * HWc) / 8, 256, 0, stream>>>(RT, topv, topi);

    // 6) gumbel k selection, 7) masked sigmoid weights per level
    kselect_kernel<<<1, 32, 0, stream>>>(klog, temp, khard);
    wsel_kernel<<<blk((size_t)3 * NB * HWc), 256, 0, stream>>>(topv, khard, wsel);

    // 8) transfer per level (lvl0=T3, lvl1=T2, lvl2=T1)
    transfer_kernel<<<blk((size_t)NB*256*48*48), 256, 0, stream>>>(
        enc_ref3, wsel + (size_t)0 * NB * 5 * HWc, topi, T3, 256, 1, 3, 1);
    transfer_kernel<<<blk((size_t)NB*128*96*96), 256, 0, stream>>>(
        enc_ref2, wsel + (size_t)1 * NB * 5 * HWc, topi, T2, 128, 2, 6, 2);
    transfer_kernel<<<blk((size_t)NB*64*192*192), 256, 0, stream>>>(
        enc_ref1, wsel + (size_t)2 * NB * 5 * HWc, topi, T1, 64, 4, 12, 4);

    // 9) squeeze-excite
    se_fc_kernel<<<NB, 256, 0, stream>>>(T3, se3w1, se3b1, se3w2, se3b2, sesc3, 256, 48*48);
    se_fc_kernel<<<NB, 256, 0, stream>>>(T2, se2w1, se2b1, se2w2, se2b2, sesc2, 128, 96*96);
    se_fc_kernel<<<NB, 256, 0, stream>>>(T1, se1w1, se1b1, se1w2, se1b2, sesc1, 64, 192*192);
    se_scale_kernel<<<blk((size_t)NB*256*48*48), 256, 0, stream>>>(T3, sesc3, 256, 48*48);
    se_scale_kernel<<<blk((size_t)NB*128*96*96), 256, 0, stream>>>(T2, sesc2, 128, 96*96);
    se_scale_kernel<<<blk((size_t)NB*64*192*192), 256, 0, stream>>>(T1, sesc1, 64, 192*192);
}
